// SegLossModule_28209345200590
// MI455X (gfx1250) — compile-verified, matrix operand order checked by reading, not run on hardware
//
#include <hip/hip_runtime.h>
#include <hip/hip_bf16.h>
#include <math.h>

typedef __attribute__((ext_vector_type(16))) _Float16 v16h;
typedef __attribute__((ext_vector_type(8)))  float    v8f;

#define B_   8
#define T_   300
#define S_   20
#define V_   2000
#define L_   30
#define VP_  2016          // padded row stride: 63 K-chunks of 32, no tail
#define VP4_ (VP_ / 4)     // 504 float4 per padded row
#define V4_  (V_ / 4)      // 500 float4 of real data per row
#define NEGV  (-1e30f)
#define LOG2E 1.4426950408889634f
#define LN2   0.6931471805599453f

// exp(x - rm) in log2 domain: one v_fma + one v_exp per element.
__device__ __forceinline__ _Float16 expdif(float x, float nrml) {
  return (_Float16)__builtin_amdgcn_exp2f(fmaf(x, LOG2E, nrml));
}

// ---------------------------------------------------------------------------
// Stage 1: one block per (b, t).  Stream the 20x2000 fp32 slab into a padded
// LDS tile once, compute per-row max, then rowsum = exp(row - max) via WMMA
// against a ones matrix (matrix-pipe row reduction).  All hot-loop accesses
// are statically in-bounds: no predication, b128 LDS traffic only.
// ---------------------------------------------------------------------------
__global__ __launch_bounds__(256) void seg_scores_kernel(
    const float* __restrict__ W, const int* __restrict__ labels,
    float* __restrict__ Wv, float* __restrict__ Wlab)
{
  extern __shared__ float smem[];
  float* tile   = smem;             // S_ * VP_ floats (padded rows)
  float* rowmax = smem + S_ * VP_;  // 32
  float* rowsum = rowmax + 32;      // 32

  const int blk  = blockIdx.x;      // blk = b*T_ + t
  const int b    = blk / T_;
  const int tid  = threadIdx.x;
  const int lane = tid & 31;
  const int wave = tid >> 5;        // 8 waves

  // --- stream global -> LDS (single HBM pass over W), b128 both sides ---
  const float4* src4 = (const float4*)(W + (size_t)blk * (S_ * V_));
  float4*       dst4 = (float4*)tile;
  for (int r = 0; r < S_; ++r) {
    const float4* srow = src4 + r * V4_;
    float4*       drow = dst4 + r * VP4_;
    for (int j = tid; j < V4_; j += 256) {
      __builtin_prefetch(&srow[j + 512], 0, 0);    // global_prefetch_b8
      drow[j] = srow[j];
    }
  }
  // fill column pad [2000,2016) with -1e30 so exp() of pad == 0
  if (tid < S_ * 4) {
    const int s = tid >> 2, k = tid & 3;
    dst4[s * VP4_ + V4_ + k] = make_float4(NEGV, NEGV, NEGV, NEGV);
  }
  if (tid < 32) { rowsum[tid] = 0.0f; if (tid >= S_) rowmax[tid] = 0.0f; }
  __syncthreads();

  // --- per-row maxima (wave-per-row, float4 LDS reads, shuffle reduce) ---
  for (int s = wave; s < S_; s += 8) {
    float m = NEGV;
    for (int i = lane; i < VP4_; i += 32) {        // pad is NEGV: harmless
      const float4 v = dst4[s * VP4_ + i];
      m = fmaxf(m, fmaxf(fmaxf(v.x, v.y), fmaxf(v.z, v.w)));
    }
    for (int off = 16; off; off >>= 1) m = fmaxf(m, __shfl_xor(m, off, 32));
    if (lane == 0) rowmax[s] = m;
  }
  __syncthreads();

  // --- rowsum of exp(w - max) via v_wmma_f32_16x16x32_f16, B = ones ---
  const int m0   = (wave & 1) * 16;                // M tile
  const int row  = m0 + (lane & 15);
  const int srow = row < S_ ? row : (S_ - 1);      // clamp: dup rows ignored
  const int hi   = lane >> 4;                      // lane half -> K sub-block
  const float nrml = -rowmax[srow] * LOG2E;        // -max in log2 domain
  const float4* trow = dst4 + srow * VP4_;

  v16h ones;
  #pragma unroll
  for (int e = 0; e < 16; ++e) ones[e] = (_Float16)1.0f;

  v8f acc = {};
  for (int kc = (wave >> 1); kc < VP_ / 32; kc += 4) {
    const int c4 = kc * 8 + hi * 2;                // float4 index of K base
    const float4 p0 = trow[c4];                    // K locals 0..3
    const float4 p1 = trow[c4 + 1];                // K locals 4..7
    const float4 q0 = trow[c4 + 4];                // K locals 16..19
    const float4 q1 = trow[c4 + 5];                // K locals 20..23
    v16h a;
    a[0]  = expdif(p0.x, nrml);
    a[1]  = expdif(p0.y, nrml);
    a[2]  = expdif(p0.z, nrml);
    a[3]  = expdif(p0.w, nrml);
    a[4]  = expdif(p1.x, nrml);
    a[5]  = expdif(p1.y, nrml);
    a[6]  = expdif(p1.z, nrml);
    a[7]  = expdif(p1.w, nrml);
    a[8]  = expdif(q0.x, nrml);
    a[9]  = expdif(q0.y, nrml);
    a[10] = expdif(q0.z, nrml);
    a[11] = expdif(q0.w, nrml);
    a[12] = expdif(q1.x, nrml);
    a[13] = expdif(q1.y, nrml);
    a[14] = expdif(q1.z, nrml);
    a[15] = expdif(q1.w, nrml);
    acc = __builtin_amdgcn_wmma_f32_16x16x32_f16(false, a, false, ones,
                                                 (short)0, acc, false, false);
  }
  // D[m][n] == rowsum[m] for all n; merge the 4 wave-partials per M tile.
  // rowsum has 32 slots and slots >= S_ are never read, so no guards needed.
  if ((lane & 15) == 0) {
    #pragma unroll
    for (int r = 0; r < 8; ++r)
      atomicAdd(&rowsum[m0 + r + hi * 8], acc[r]);  // ds_add_f32, unpredicated
  }
  __syncthreads();

  if (tid < S_)
    Wv[(size_t)blk * S_ + tid] =
        rowmax[tid] + __builtin_amdgcn_logf(rowsum[tid]) * LN2;

  for (int i = tid; i < S_ * L_; i += 256) {
    const int s = i / L_, l = i - s * L_;
    const int lab = labels[b * L_ + l];
    Wlab[((size_t)blk * S_ + s) * L_ + l] = tile[s * VP_ + lab];
  }
}

// ---------------------------------------------------------------------------
// Stage 2: sequential DP over t = 1..T with circular alpha ring buffers in
// LDS.  One block, 256 threads: tid < 240 -> (b,l) numerator cells,
// tid in [240,248) -> denominator cells.
// ---------------------------------------------------------------------------
__global__ __launch_bounds__(256) void seg_dp_kernel(
    const float* __restrict__ Wv, const float* __restrict__ Wlab,
    const int* __restrict__ label_sizes, const int* __restrict__ prob_sizes,
    const int* __restrict__ is_sum, float* __restrict__ out)
{
  __shared__ float bf[S_][B_];            // alpha_free[t-d] at slot (t-d)%S
  __shared__ float bn[S_][B_][L_ + 1];    // alpha_num
  __shared__ float logZ[B_], lognum[B_];
  __shared__ int   ps[B_], ls[B_];

  const int tid = threadIdx.x;

  for (int i = tid; i < S_ * B_; i += 256) {
    const int d = i / B_, b = i % B_;
    bf[d][b] = (d == 0) ? 0.0f : NEGV;    // alpha[0] = 0
  }
  for (int i = tid; i < S_ * B_ * (L_ + 1); i += 256) {
    const int d = i / (B_ * (L_ + 1));
    const int rem = i - d * (B_ * (L_ + 1));
    const int b = rem / (L_ + 1), l = rem % (L_ + 1);
    bn[d][b][l] = (d == 0 && l == 0) ? 0.0f : NEGV;
  }
  if (tid < B_) {
    ps[tid] = prob_sizes[tid];
    ls[tid] = label_sizes[tid];
    logZ[tid]   = 0.0f;                                  // free_all[0] = 0
    lognum[tid] = (label_sizes[tid] == 0) ? 0.0f : NEGV; // num_all[0]
  }
  __syncthreads();

  for (int t = 1; t <= T_; ++t) {
    float val = NEGV;
    int wb = -1, wl = 0;

    if (tid < B_ * L_) {                          // numerator cell (b, l)
      const int b = tid / L_, l = tid - (tid / L_) * L_;
      float terms[S_], mx = NEGV;
      #pragma unroll
      for (int d = 0; d < S_; ++d) {
        const int start = t - (d + 1);
        const float x = (start >= 0)
            ? Wlab[((size_t)(b * T_ + start) * S_ + d) * L_ + l] : NEGV;
        const float a = bn[(start + S_) % S_][b][l];
        terms[d] = a + x;
        mx = fmaxf(mx, terms[d]);
      }
      float sum = 0.0f;
      #pragma unroll
      for (int d = 0; d < S_; ++d) sum += __expf(terms[d] - mx);
      val = mx + __logf(sum);
      wb = b; wl = l + 1;
    } else if (tid < B_ * L_ + B_) {              // denominator cell (b)
      const int b = tid - B_ * L_;
      float terms[S_], mx = NEGV;
      #pragma unroll
      for (int d = 0; d < S_; ++d) {
        const int start = t - (d + 1);
        const float x = (start >= 0)
            ? Wv[(size_t)(b * T_ + start) * S_ + d] : NEGV;
        terms[d] = bf[(start + S_) % S_][b] + x;
        mx = fmaxf(mx, terms[d]);
      }
      float sum = 0.0f;
      #pragma unroll
      for (int d = 0; d < S_; ++d) sum += __expf(terms[d] - mx);
      val = mx + __logf(sum);
      wb = b;
    }
    __syncthreads();                              // all reads done

    const int slot = t % S_;
    if (tid < B_ * L_) {
      bn[slot][wb][wl] = val;
      if (t == ps[wb] && wl == ls[wb]) lognum[wb] = val;
    } else if (tid < B_ * L_ + B_) {
      bf[slot][wb] = val;
      bn[slot][wb][0] = NEGV;                     // nn_full[:,0] = NEG
      if (t == ps[wb]) logZ[wb] = val;
    }
    __syncthreads();                              // writes visible next step
  }

  if (tid == 0) {
    float s = 0.0f;
    for (int b = 0; b < B_; ++b) s += logZ[b] - lognum[b];
    if (is_sum[0] == 0) s /= (float)B_;
    out[0] = s;
  }
}

// ---------------------------------------------------------------------------
extern "C" void kernel_launch(void* const* d_in, const int* in_sizes, int n_in,
                              void* d_out, int out_size, void* d_ws, size_t ws_size,
                              hipStream_t stream) {
  (void)in_sizes; (void)n_in; (void)out_size; (void)ws_size;
  const float* W      = (const float*)d_in[0];
  const int*   labels = (const int*)d_in[1];
  const int*   lsz    = (const int*)d_in[2];
  const int*   psz    = (const int*)d_in[3];
  const int*   issum  = (const int*)d_in[4];
  float*       out    = (float*)d_out;

  float* Wv   = (float*)d_ws;                       // B*T*S   = 48000 floats
  float* Wlab = Wv + (size_t)B_ * T_ * S_;          // B*T*S*L = 1.44M floats

  const size_t smem = (size_t)(S_ * VP_ + 64) * sizeof(float);  // ~161.5 KB
  seg_scores_kernel<<<B_ * T_, 256, smem, stream>>>(W, labels, Wv, Wlab);
  seg_dp_kernel<<<1, 256, 0, stream>>>(Wv, Wlab, lsz, psz, issum, out);
}
